// CanonicalMLP_9302899163588
// MI455X (gfx1250) — matrix operand order repaired and checked by reading, not compiled, not run on hardware
//
#include <hip/hip_runtime.h>
#include <hip/hip_bf16.h>
#include <hip/hip_fp16.h>
#include <math.h>

// ---------------------------------------------------------------------------
// Types / constants
// ---------------------------------------------------------------------------
typedef _Float16 h8   __attribute__((ext_vector_type(8)));
typedef _Float16 v16h __attribute__((ext_vector_type(16)));
typedef float    v8f  __attribute__((ext_vector_type(8)));

constexpr int BATCH = 32;
constexpr int NPTS  = 1024;
constexpr int KNB   = 20;      // k of knn
constexpr float BN_EPS = 1e-5f;
constexpr float NEG_SLOPE = 0.2f;

#define DEVFN __device__ __forceinline__

// ---------------------------------------------------------------------------
// Small math helpers
// ---------------------------------------------------------------------------
DEVFN float wave_sum32(float v) {
#pragma unroll
  for (int o = 16; o > 0; o >>= 1) v += __shfl_xor(v, o, 32);
  return v;
}

DEVFN bool lex_gt(float ax, float ay, float az, float bx, float by, float bz) {
  if (ax != bx) return ax > bx;
  if (ay != by) return ay > by;
  return az > bz;
}

DEVFN void cross3(const float* u, const float* v, float* w) {
  w[0] = u[1]*v[2] - u[2]*v[1];
  w[1] = u[2]*v[0] - u[0]*v[2];
  w[2] = u[0]*v[1] - u[1]*v[0];
}

DEVFN void eigvec_for(float a00, float a01, float a02, float a11, float a12,
                      float a22, float lam, float* v) {
  float r0[3] = {a00 - lam, a01, a02};
  float r1[3] = {a01, a11 - lam, a12};
  float r2[3] = {a02, a12, a22 - lam};
  float c0[3], c1[3], c2[3];
  cross3(r0, r1, c0); cross3(r0, r2, c1); cross3(r1, r2, c2);
  float n0 = c0[0]*c0[0] + c0[1]*c0[1] + c0[2]*c0[2];
  float n1 = c1[0]*c1[0] + c1[1]*c1[1] + c1[2]*c1[2];
  float n2 = c2[0]*c2[0] + c2[1]*c2[1] + c2[2]*c2[2];
  float* best = c0; float nb = n0;
  if (n1 > nb) { best = c1; nb = n1; }
  if (n2 > nb) { best = c2; nb = n2; }
  if (nb < 1e-24f) { v[0] = 1.f; v[1] = 0.f; v[2] = 0.f; return; }
  float inv = rsqrtf(nb);
  v[0] = best[0] * inv; v[1] = best[1] * inv; v[2] = best[2] * inv;
}

// columns of V (V[row*3+col]) = eigenvectors, descending eigenvalue
DEVFN void eig3_desc(float a00, float a01, float a02, float a11, float a12,
                     float a22, float* V) {
  float p1 = a01*a01 + a02*a02 + a12*a12;
  float q  = (a00 + a11 + a22) * (1.f/3.f);
  float d0 = a00 - q, d1 = a11 - q, d2 = a22 - q;
  float p2 = d0*d0 + d1*d1 + d2*d2 + 2.f*p1;
  float p  = sqrtf(fmaxf(p2 * (1.f/6.f), 1e-30f));
  float ip = 1.f / p;
  float b00 = d0*ip, b11 = d1*ip, b22 = d2*ip;
  float b01 = a01*ip, b02 = a02*ip, b12 = a12*ip;
  float detB = b00*(b11*b22 - b12*b12) - b01*(b01*b22 - b12*b02)
             + b02*(b01*b12 - b11*b02);
  float r   = fminf(fmaxf(0.5f*detB, -1.f), 1.f);
  float phi = acosf(r) * (1.f/3.f);
  float l0 = q + 2.f*p*cosf(phi);
  float l2 = q + 2.f*p*cosf(phi + 2.0943951023931953f);
  float l1 = 3.f*q - l0 - l2;
  float v0[3], v1[3], v2[3];
  eigvec_for(a00,a01,a02,a11,a12,a22, l0, v0);
  eigvec_for(a00,a01,a02,a11,a12,a22, l1, v1);
  eigvec_for(a00,a01,a02,a11,a12,a22, l2, v2);
  V[0]=v0[0]; V[3]=v0[1]; V[6]=v0[2];
  V[1]=v1[0]; V[4]=v1[1]; V[7]=v1[2];
  V[2]=v2[0]; V[5]=v2[1]; V[8]=v2[2];
}

// reference sign canonicalization: per-column argmax(|v|) sign, then det fix
DEVFN void fix_signs(float* V) {
#pragma unroll
  for (int c = 0; c < 3; ++c) {
    float a0 = fabsf(V[c]), a1 = fabsf(V[3+c]), a2 = fabsf(V[6+c]);
    int im = 0; float am = a0;
    if (a1 > am) { am = a1; im = 1; }
    if (a2 > am) { am = a2; im = 2; }
    float s = (V[im*3 + c] < 0.f) ? -1.f : 1.f;
    V[c] *= s; V[3+c] *= s; V[6+c] *= s;
  }
  float det = V[0]*(V[4]*V[8] - V[5]*V[7]) - V[1]*(V[3]*V[8] - V[5]*V[6])
            + V[2]*(V[3]*V[7] - V[4]*V[6]);
  if (det < 0.f) { V[2] = -V[2]; V[5] = -V[5]; V[8] = -V[8]; }
}

DEVFN float lrelu(float v) { return fmaxf(v, 0.f) + NEG_SLOPE * fminf(v, 0.f); }

// WMMA fragment loaders (16-bit A 16x32 / B 32x16 lane layouts, wave32)
DEVFN v16h load_frag_a(const _Float16* p) {     // p = W + row*Cp + k0 + akb
  h8 lo = *(const h8*)(p);
  h8 hi = *(const h8*)(p + 16);
  return __builtin_shufflevector(lo, hi, 0,1,2,3,4,5,6,7,8,9,10,11,12,13,14,15);
}
DEVFN v16h load_frag_b(const _Float16* p) {     // p = X + row*stride + k0 + bkb
  h8 lo = *(const h8*)(p);
  h8 hi = *(const h8*)(p + 16);
  return __builtin_shufflevector(lo, hi, 0,1,2,3,4,5,6,7,8,9,10,11,12,13,14,15);
}

// async global->LDS b128 copy (gfx1250 ASYNCcnt path)
DEVFN void async_cp_b128(unsigned lds_off, const _Float16* gptr) {
  asm volatile("global_load_async_to_lds_b128 %0, %1, off"
               :: "v"(lds_off), "v"(gptr) : "memory");
}

// ---------------------------------------------------------------------------
// bn fold:  scale = g*rsqrt(v+eps), shift = (bias - m)*scale + b
// ---------------------------------------------------------------------------
__global__ void bn_prep_kernel(const float* __restrict__ bn,
                               const float* __restrict__ bias,
                               float* __restrict__ scale,
                               float* __restrict__ shift, int C) {
  int c = blockIdx.x * 256 + threadIdx.x;
  if (c >= C) return;
  float g = bn[c], b = bn[C + c], m = bn[2*C + c], v = bn[3*C + c];
  float s = g * rsqrtf(v + BN_EPS);
  float bi = bias ? bias[c] : 0.f;
  scale[c] = s;
  shift[c] = (bi - m) * s + b;
}

// weight f32 [O][Cc] -> f16 [O][Cp] zero padded
__global__ void wconv_kernel(const float* __restrict__ W,
                             _Float16* __restrict__ Wh,
                             int Cc, int Cp, long total) {
  long i = (long)blockIdx.x * 256 + threadIdx.x;
  if (i >= total) return;
  long o = i / Cp;
  int  c = (int)(i - o * Cp);
  Wh[i] = (c < Cc) ? (_Float16)W[o * Cc + c] : (_Float16)0.f;
}

__global__ void cvt_f32_to_f16(const float* __restrict__ src,
                               _Float16* __restrict__ dst, long n) {
  long i = (long)blockIdx.x * 256 + threadIdx.x;
  if (i < n) dst[i] = (_Float16)src[i];
}

// ---------------------------------------------------------------------------
// Global cloud PCA + skew + lexsort.  One block (512 thr) per cloud.
// in: x [B][3][N] f32   out: pts [B][N][3] f32 (sorted canonical)
// ---------------------------------------------------------------------------
__global__ __launch_bounds__(512) void global_pca_kernel(
    const float* __restrict__ x, float* __restrict__ pts) {
  __shared__ float red[12];
  __shared__ float V[9];
  __shared__ float sgn[3];
  __shared__ float sx[NPTS], sy[NPTS], sz[NPTS];
  const int b = blockIdx.x;
  const int t = threadIdx.x;
  if (t < 12) red[t] = 0.f;
  __syncthreads();

  const float* xb = x + (size_t)b * 3 * NPTS;
  float px[2], py[2], pz[2];
#pragma unroll
  for (int i = 0; i < 2; ++i) {
    int n = t + i * 512;
    px[i] = xb[n]; py[i] = xb[NPTS + n]; pz[i] = xb[2*NPTS + n];
  }
  atomicAdd(&red[0], px[0] + px[1]);
  atomicAdd(&red[1], py[0] + py[1]);
  atomicAdd(&red[2], pz[0] + pz[1]);
  __syncthreads();
  const float mx = red[0] / NPTS, my = red[1] / NPTS, mz = red[2] / NPTS;

  float cxx = 0, cxy = 0, cxz = 0, cyy = 0, cyz = 0, czz = 0;
#pragma unroll
  for (int i = 0; i < 2; ++i) {
    float cx = px[i] - mx, cy = py[i] - my, cz = pz[i] - mz;
    cxx += cx*cx; cxy += cx*cy; cxz += cx*cz;
    cyy += cy*cy; cyz += cy*cz; czz += cz*cz;
  }
  atomicAdd(&red[3], cxx); atomicAdd(&red[4], cxy); atomicAdd(&red[5], cxz);
  atomicAdd(&red[6], cyy); atomicAdd(&red[7], cyz); atomicAdd(&red[8], czz);
  __syncthreads();

  if (t == 0) {
    const float inv = 1.f / (NPTS - 1);
    float Vl[9];
    eig3_desc(red[3]*inv, red[4]*inv, red[5]*inv,
              red[6]*inv, red[7]*inv, red[8]*inv, Vl);
    fix_signs(Vl);
#pragma unroll
    for (int i = 0; i < 9; ++i) V[i] = Vl[i];
  }
  __syncthreads();

  float c3[3] = {0.f, 0.f, 0.f};
#pragma unroll
  for (int i = 0; i < 2; ++i) {
    int n = t + i * 512;
    float cx = px[i] - mx, cy = py[i] - my, cz = pz[i] - mz;
    float q0 = cx*V[0] + cy*V[3] + cz*V[6];
    float q1 = cx*V[1] + cy*V[4] + cz*V[7];
    float q2 = cx*V[2] + cy*V[5] + cz*V[8];
    sx[n] = q0; sy[n] = q1; sz[n] = q2;
    c3[0] += q0*q0*q0; c3[1] += q1*q1*q1; c3[2] += q2*q2*q2;
  }
  atomicAdd(&red[9], c3[0]); atomicAdd(&red[10], c3[1]); atomicAdd(&red[11], c3[2]);
  __syncthreads();
  if (t == 0) {
    float s0 = (red[9]  < 0.f) ? -1.f : 1.f;
    float s1 = (red[10] < 0.f) ? -1.f : 1.f;
    float s2 = (red[11] < 0.f) ? -1.f : 1.f;
    int neg = (s0 < 0.f) + (s1 < 0.f) + (s2 < 0.f);
    if (neg & 1) s2 = -s2;
    sgn[0] = s0; sgn[1] = s1; sgn[2] = s2;
  }
  __syncthreads();
#pragma unroll
  for (int i = 0; i < 2; ++i) {
    int n = t + i * 512;
    sx[n] *= sgn[0]; sy[n] *= sgn[1]; sz[n] *= sgn[2];
  }

  for (int ksz = 2; ksz <= NPTS; ksz <<= 1) {
    for (int jsz = ksz >> 1; jsz > 0; jsz >>= 1) {
      __syncthreads();
#pragma unroll
      for (int rep = 0; rep < 2; ++rep) {
        int i = t + rep * 512;
        int ixj = i ^ jsz;
        if (ixj > i) {
          bool up = ((i & ksz) == 0);
          float ax = sx[i], ay = sy[i], az = sz[i];
          float bx = sx[ixj], by = sy[ixj], bz = sz[ixj];
          if (up == lex_gt(ax, ay, az, bx, by, bz)) {
            sx[i] = bx; sy[i] = by; sz[i] = bz;
            sx[ixj] = ax; sy[ixj] = ay; sz[ixj] = az;
          }
        }
      }
    }
  }
  __syncthreads();
  float* pb = pts + (size_t)b * NPTS * 3;
#pragma unroll
  for (int i = 0; i < 2; ++i) {
    int n = t + i * 512;
    pb[n*3 + 0] = sx[n]; pb[n*3 + 1] = sy[n]; pb[n*3 + 2] = sz[n];
  }
}

// ---------------------------------------------------------------------------
// Scalar KNN top-20 (stage 1, C=3 only — negligible FLOPs).
// ---------------------------------------------------------------------------
__global__ __launch_bounds__(64) void knn_topk_kernel(
    const float* __restrict__ F, int* __restrict__ idx, int C) {
  const int b = blockIdx.y;
  const int n = blockIdx.x * 64 + threadIdx.x;
  const float* fn = F + ((size_t)b * NPTS + n) * C;
  float best[KNB]; int bi[KNB];
#pragma unroll
  for (int j = 0; j < KNB; ++j) { best[j] = -3.4e38f; bi[j] = 0; }
  for (int m = 0; m < NPTS; ++m) {
    const float* fm = F + ((size_t)b * NPTS + m) * C;
    float d = 0.f;
    for (int c = 0; c < C; ++c) {
      float t = fn[c] - fm[c];
      d += t * t;
    }
    float pd = -d;
    if (pd > best[KNB - 1]) {
      int pos = KNB - 1;
      while (pos > 0 && pd > best[pos - 1]) {
        best[pos] = best[pos - 1]; bi[pos] = bi[pos - 1]; --pos;
      }
      best[pos] = pd; bi[pos] = m;
    }
  }
  int* op = idx + ((size_t)b * NPTS + n) * KNB;
#pragma unroll
  for (int j = 0; j < KNB; ++j) op[j] = bi[j];
}

// ---------------------------------------------------------------------------
// WMMA Gram matrix for KNN (stages 2-4):  G[b] = F[b] * F[b]^T, f16 in f32 acc
// ---------------------------------------------------------------------------
__global__ __launch_bounds__(256) void gram_wmma_kernel(
    const _Float16* __restrict__ F, float* __restrict__ G, int Cp) {
  const int lane = threadIdx.x & 31;
  const int wave = threadIdx.x >> 5;
  const int gn = NPTS >> 7;                 // 8
  const int bm = blockIdx.x / gn;
  const int bn = blockIdx.x - bm * gn;
  const int wm = wave >> 2, wn = wave & 3;
  const int lr = lane & 15;
  const int akb = (lane >> 4) * 8;
  const int bkb = (lane >> 4) * 8;
  const _Float16* Fb = F + (size_t)blockIdx.y * NPTS * Cp;
  const _Float16* ap0 = Fb + (size_t)(bm*64  + wm*32 + lr) * Cp;
  const _Float16* ap1 = ap0 + (size_t)16 * Cp;
  const _Float16* bp0 = Fb + (size_t)(bn*128 + wn*32 + lr) * Cp;
  const _Float16* bp1 = bp0 + (size_t)16 * Cp;

  v8f a00 = {}, a01 = {}, a10 = {}, a11 = {};
  for (int k0 = 0; k0 < Cp; k0 += 32) {
    v16h af0 = load_frag_a(ap0 + k0 + akb);
    v16h af1 = load_frag_a(ap1 + k0 + akb);
    v16h bf0 = load_frag_b(bp0 + k0 + bkb);
    v16h bf1 = load_frag_b(bp1 + k0 + bkb);
    a00 = __builtin_amdgcn_wmma_f32_16x16x32_f16(false, af0, false, bf0, (short)0, a00, false, false);
    a01 = __builtin_amdgcn_wmma_f32_16x16x32_f16(false, af0, false, bf1, (short)0, a01, false, false);
    a10 = __builtin_amdgcn_wmma_f32_16x16x32_f16(false, af1, false, bf0, (short)0, a10, false, false);
    a11 = __builtin_amdgcn_wmma_f32_16x16x32_f16(false, af1, false, bf1, (short)0, a11, false, false);
  }
  float* Gb = G + (size_t)blockIdx.y * NPTS * NPTS;
#pragma unroll
  for (int tm = 0; tm < 2; ++tm) {
    const int gm = bm*64 + wm*32 + tm*16 + (lane >> 4) * 8;
#pragma unroll
    for (int tn = 0; tn < 2; ++tn) {
      const int gc = bn*128 + wn*32 + tn*16 + lr;
      v8f a = (tm == 0) ? (tn == 0 ? a00 : a01) : (tn == 0 ? a10 : a11);
#pragma unroll
      for (int r = 0; r < 8; ++r)
        Gb[(size_t)(gm + r) * NPTS + gc] = a[r];
    }
  }
}

__global__ void extract_diag_kernel(const float* __restrict__ G,
                                    float* __restrict__ diag) {
  long i = (long)blockIdx.x * 256 + threadIdx.x;   // over B*N
  long b = i >> 10;
  int  n = (int)(i & 1023);
  diag[i] = G[(size_t)b * NPTS * NPTS + (size_t)n * NPTS + n];
}

// top-20 of pd[m] = 2*G[n][m] - G[n][n] - G[m][m]
__global__ __launch_bounds__(64) void topk_gram_kernel(
    const float* __restrict__ G, const float* __restrict__ diag,
    int* __restrict__ idx) {
  const int b = blockIdx.y;
  const int n = blockIdx.x * 64 + threadIdx.x;
  const float dn = diag[b * NPTS + n];
  const float* g  = G + ((size_t)b * NPTS + n) * NPTS;
  const float* dm = diag + b * NPTS;
  float best[KNB]; int bi[KNB];
#pragma unroll
  for (int j = 0; j < KNB; ++j) { best[j] = -3.4e38f; bi[j] = 0; }
  for (int m = 0; m < NPTS; ++m) {
    float pd = 2.f * g[m] - dn - dm[m];
    if (pd > best[KNB - 1]) {
      int pos = KNB - 1;
      while (pos > 0 && pd > best[pos - 1]) {
        best[pos] = best[pos - 1]; bi[pos] = bi[pos - 1]; --pos;
      }
      best[pos] = pd; bi[pos] = m;
    }
  }
  int* op = idx + ((size_t)b * NPTS + n) * KNB;
#pragma unroll
  for (int j = 0; j < KNB; ++j) op[j] = bi[j];
}

// ---------------------------------------------------------------------------
// Per-patch PCA + skew + lexsort. One wave (32 lanes) per patch; 8 patches/blk
// ---------------------------------------------------------------------------
__global__ __launch_bounds__(256) void patch_pca_kernel(
    const float* __restrict__ pts, const int* __restrict__ idx,
    float* __restrict__ canon, int* __restrict__ perm) {
  __shared__ float lsx[8][KNB], lsy[8][KNB], lsz[8][KNB];
  __shared__ int   lsp[8][KNB];
  const int lane = threadIdx.x & 31;
  const int wave = threadIdx.x >> 5;
  const int patch = blockIdx.x * 8 + wave;   // = b*NPTS + n
  const int b = patch >> 10;

  float px = 0.f, py = 0.f, pz = 0.f;
  const float wmask = (lane < KNB) ? 1.f : 0.f;
  if (lane < KNB) {
    int g = idx[(size_t)patch * KNB + lane];
    const float* p = pts + ((size_t)b * NPTS + g) * 3;
    px = p[0]; py = p[1]; pz = p[2];
  }
  const float mx = wave_sum32(wmask * px) * (1.f / KNB);
  const float my = wave_sum32(wmask * py) * (1.f / KNB);
  const float mz = wave_sum32(wmask * pz) * (1.f / KNB);
  const float cx = px - mx, cy = py - my, cz = pz - mz;
  const float inv = 1.f / (KNB - 1);
  float a00 = wave_sum32(wmask * cx * cx) * inv;
  float a01 = wave_sum32(wmask * cx * cy) * inv;
  float a02 = wave_sum32(wmask * cx * cz) * inv;
  float a11 = wave_sum32(wmask * cy * cy) * inv;
  float a12 = wave_sum32(wmask * cy * cz) * inv;
  float a22 = wave_sum32(wmask * cz * cz) * inv;

  float V[9];
  eig3_desc(a00, a01, a02, a11, a12, a22, V);
  fix_signs(V);

  float q0 = cx*V[0] + cy*V[3] + cz*V[6];
  float q1 = cx*V[1] + cy*V[4] + cz*V[7];
  float q2 = cx*V[2] + cy*V[5] + cz*V[8];

  float m0 = wave_sum32(wmask * q0*q0*q0);
  float m1 = wave_sum32(wmask * q1*q1*q1);
  float m2 = wave_sum32(wmask * q2*q2*q2);
  float s0 = (m0 < 0.f) ? -1.f : 1.f;
  float s1 = (m1 < 0.f) ? -1.f : 1.f;
  float s2 = (m2 < 0.f) ? -1.f : 1.f;
  int neg = (s0 < 0.f) + (s1 < 0.f) + (s2 < 0.f);
  if (neg & 1) s2 = -s2;
  q0 *= s0; q1 *= s1; q2 *= s2;

  if (lane < KNB) {
    lsx[wave][lane] = q0; lsy[wave][lane] = q1; lsz[wave][lane] = q2;
    lsp[wave][lane] = lane;
  }
  __syncthreads();
  if (lane == 0) {
    for (int a = 1; a < KNB; ++a) {
      float vx = lsx[wave][a], vy = lsy[wave][a], vz = lsz[wave][a];
      int vp = lsp[wave][a];
      int p = a - 1;
      while (p >= 0 &&
             lex_gt(lsx[wave][p], lsy[wave][p], lsz[wave][p], vx, vy, vz)) {
        lsx[wave][p+1] = lsx[wave][p]; lsy[wave][p+1] = lsy[wave][p];
        lsz[wave][p+1] = lsz[wave][p]; lsp[wave][p+1] = lsp[wave][p];
        --p;
      }
      lsx[wave][p+1] = vx; lsy[wave][p+1] = vy; lsz[wave][p+1] = vz;
      lsp[wave][p+1] = vp;
    }
  }
  __syncthreads();
  if (lane < KNB) {
    float* cp = canon + (size_t)patch * KNB * 3 + lane * 3;
    cp[0] = lsx[wave][lane]; cp[1] = lsy[wave][lane]; cp[2] = lsz[wave][lane];
    perm[(size_t)patch * KNB + lane] = lsp[wave][lane];
  }
}

// ---------------------------------------------------------------------------
// Patch feature builders (point-major f16, zero-padded to Cp)
// ---------------------------------------------------------------------------
__global__ void build_p1_kernel(const float* __restrict__ canon,
                                _Float16* __restrict__ P) {
  long i = (long)blockIdx.x * 256 + threadIdx.x;   // over B*N*64
  int f = (int)(i & 63);
  long pnt = i >> 6;
  P[i] = (f < 60) ? (_Float16)canon[pnt * 60 + f] : (_Float16)0.f;
}

__global__ void build_pf_kernel(const float* __restrict__ canon,
                                const int* __restrict__ idx,
                                const int* __restrict__ perm,
                                const float* __restrict__ xprev,
                                _Float16* __restrict__ P, int C, int Cp) {
  const long pnt = blockIdx.x;            // 0..B*N-1
  const long b = pnt >> 10;
  const int S = 3 + C;
  const int used = KNB * S;
  for (int f = threadIdx.x; f < Cp; f += blockDim.x) {
    float v = 0.f;
    if (f < used) {
      int j = f / S;
      int r = f - j * S;
      if (r < 3) {
        v = canon[pnt * (KNB * 3) + j * 3 + r];
      } else {
        int g = idx[pnt * KNB + perm[pnt * KNB + j]];
        v = xprev[((size_t)b * NPTS + g) * C + (r - 3)];
      }
    }
    P[pnt * Cp + f] = (_Float16)v;
  }
}

__global__ void build_xc_kernel(const float* __restrict__ x1,
                                const float* __restrict__ x2,
                                const float* __restrict__ x3,
                                const float* __restrict__ x4,
                                _Float16* __restrict__ xc) {
  long i = (long)blockIdx.x * 256 + threadIdx.x;   // over B*N*512
  int c = (int)(i & 511);
  long p = i >> 9;
  float v;
  if      (c < 64)  v = x1[p * 64  + c];
  else if (c < 128) v = x2[p * 64  + (c - 64)];
  else if (c < 256) v = x3[p * 128 + (c - 128)];
  else              v = x4[p * 256 + (c - 256)];
  xc[i] = (_Float16)v;
}

// ---------------------------------------------------------------------------
// Register-blocked WMMA GEMM + BN fold + leaky ReLU (+ optional f16 copy),
// with the 128x32 B-tile double-buffered in LDS via the gfx1250 async
// global->LDS engine (ASYNCcnt).  Rows padded 32->40 halves (80B) for
// conflict-free ds_load_b128 fragment reads.
//   W : f16 [O][Cp], X : f16 [nb][Np][Cp], Y : f32 [nb][Np][O]
// Requires O % 64 == 0, Np % 128 == 0.
// ---------------------------------------------------------------------------
__global__ __launch_bounds__(256) void gemm_wmma_bn_blk_kernel(
    const _Float16* __restrict__ W, const _Float16* __restrict__ X,
    float* __restrict__ Y, _Float16* __restrict__ Yh,
    const float* __restrict__ scale, const float* __restrict__ shift,
    int O, int Cp, int Np) {
  __shared__ __align__(16) _Float16 Bt[2][128 * 40];
  const int tid  = threadIdx.x;
  const int lane = tid & 31;
  const int wave = tid >> 5;
  const int gn = Np >> 7;
  const int bm = blockIdx.x / gn;
  const int bn = blockIdx.x - bm * gn;
  const int wm = wave >> 2, wn = wave & 3;
  const int b  = blockIdx.y;
  const int lr = lane & 15;
  const int akb = (lane >> 4) * 8;
  const int bkb = (lane >> 4) * 8;

  const _Float16* wp0 = W + (size_t)(bm*64 + wm*32 + lr) * Cp;
  const _Float16* wp1 = wp0 + (size_t)16 * Cp;
  const _Float16* xblk = X + ((size_t)b * Np + bn*128) * Cp;  // block's 128 rows

  // this thread's two staging chunks (8 halves = 16B each): rows 0..127 x 4
  const int row0 = tid >> 2;            // 0..63
  const int row1 = row0 + 64;           // 64..127
  const int cc   = (tid & 3) * 8;       // K-chunk within row
  const unsigned l0a = (unsigned)(size_t)(&Bt[0][row0 * 40 + cc]);
  const unsigned l1a = (unsigned)(size_t)(&Bt[0][row1 * 40 + cc]);
  const unsigned l0b = (unsigned)(size_t)(&Bt[1][row0 * 40 + cc]);
  const unsigned l1b = (unsigned)(size_t)(&Bt[1][row1 * 40 + cc]);

  // prologue: stage K-tile 0 into buffer 0
  async_cp_b128(l0a, xblk + (size_t)row0 * Cp + cc);
  async_cp_b128(l1a, xblk + (size_t)row1 * Cp + cc);

  v8f a00 = {}, a01 = {}, a10 = {}, a11 = {};
  int buf = 0;
  for (int k0 = 0; k0 < Cp; k0 += 32) {
    const int nxt = k0 + 32;
    if (nxt < Cp) {                       // stage next tile, wait for current
      async_cp_b128(buf ? l0a : l0b, xblk + (size_t)row0 * Cp + nxt + cc);
      async_cp_b128(buf ? l1a : l1b, xblk + (size_t)row1 * Cp + nxt + cc);
      asm volatile("s_wait_asynccnt 0x2" ::: "memory");
    } else {
      asm volatile("s_wait_asynccnt 0x0" ::: "memory");
    }
    __syncthreads();

    v16h af0 = load_frag_a(wp0 + k0 + akb);
    v16h af1 = load_frag_a(wp1 + k0 + akb);
    const _Float16* bbase = &Bt[buf][0];
    v16h bf0 = load_frag_b(bbase + (wn*32 + lr) * 40 + bkb);
    v16h bf1 = load_frag_b(bbase + (wn*32 + 16 + lr) * 40 + bkb);
    a00 = __builtin_amdgcn_wmma_f32_16x16x32_f16(false, af0, false, bf0, (short)0, a00, false, false);
    a01 = __builtin_amdgcn_wmma_f32_16x16x32_f16(false, af0, false, bf1, (short)0, a01, false, false);
    a10 = __builtin_amdgcn_wmma_f32_16x16x32_f16(false, af1, false, bf0, (short)0, a10, false, false);
    a11 = __builtin_amdgcn_wmma_f32_16x16x32_f16(false, af1, false, bf1, (short)0, a11, false, false);

    __syncthreads();                      // all waves done with Bt[buf]
    buf ^= 1;
  }

#pragma unroll
  for (int tm = 0; tm < 2; ++tm) {
    const int ob = bm*64 + wm*32 + tm*16 + (lane >> 4) * 8;
    float sc[8], sh[8];
#pragma unroll
    for (int r = 0; r < 8; ++r) { sc[r] = scale[ob + r]; sh[r] = shift[ob + r]; }
#pragma unroll
    for (int tn = 0; tn < 2; ++tn) {
      const int prow = bn*128 + wn*32 + tn*16 + lr;
      v8f a = (tm == 0) ? (tn == 0 ? a00 : a01) : (tn == 0 ? a10 : a11);
      float* yp = Y + ((size_t)b * Np + prow) * O + ob;
      _Float16* yh = Yh ? Yh + ((size_t)b * Np + prow) * O + ob : (_Float16*)0;
#pragma unroll
      for (int r = 0; r < 8; ++r) {
        float v = lrelu(a[r] * sc[r] + sh[r]);
        yp[r] = v;
        if (yh) yh[r] = (_Float16)v;
      }
    }
  }
}

// ---------------------------------------------------------------------------
// Simple 1-tile/wave WMMA GEMM (FC head, M=32 rows)
// ---------------------------------------------------------------------------
__global__ __launch_bounds__(256) void gemm_wmma_bn_kernel(
    const _Float16* __restrict__ W, const _Float16* __restrict__ X,
    float* __restrict__ Y, const float* __restrict__ scale,
    const float* __restrict__ shift, int O, int Cp, int Np) {
  const int lane = threadIdx.x & 31;
  const int wave = threadIdx.x >> 5;
  const int ntN = Np >> 4;
  const int ntiles = (O >> 4) * ntN;
  const int tile = blockIdx.x * 8 + wave;
  if (tile >= ntiles) return;
  const int mt = tile / ntN;
  const int nt = tile - mt * ntN;
  const int b = blockIdx.y;

  const int akb = (lane >> 4) * 8;
  const int bkb = (lane >> 4) * 8;
  const _Float16* wp = W + (size_t)(mt * 16 + (lane & 15)) * Cp;
  const _Float16* xp = X + ((size_t)b * Np + nt * 16 + (lane & 15)) * Cp;

  v8f acc = {};
  for (int k0 = 0; k0 < Cp; k0 += 32) {
    v16h av = load_frag_a(wp + k0 + akb);
    v16h bv = load_frag_b(xp + k0 + bkb);
    acc = __builtin_amdgcn_wmma_f32_16x16x32_f16(false, av, false, bv,
                                                 (short)0, acc, false, false);
  }
  const int ob = mt * 16 + (lane >> 4) * 8;
  float* yp = Y + ((size_t)b * Np + nt * 16 + (lane & 15)) * O + ob;
#pragma unroll
  for (int r = 0; r < 8; ++r) {
    float v = acc[r] * scale[ob + r] + shift[ob + r];
    yp[r] = lrelu(v);
  }
}

// ---------------------------------------------------------------------------
// max + mean pool over N: x5 [B][N][1024] f32 -> xph [B][2048] f16
// ---------------------------------------------------------------------------
__global__ void pool_kernel(const float* __restrict__ x5,
                            _Float16* __restrict__ xph) {
  const int b = blockIdx.y;
  const int o = blockIdx.x * 256 + threadIdx.x;   // 0..1023
  float mx = -3.4e38f, sm = 0.f;
  const float* base = x5 + (size_t)b * NPTS * 1024 + o;
  for (int n = 0; n < NPTS; ++n) {
    float v = base[(size_t)n * 1024];
    mx = fmaxf(mx, v);
    sm += v;
  }
  xph[(size_t)b * 2048 + o]        = (_Float16)mx;
  xph[(size_t)b * 2048 + 1024 + o] = (_Float16)(sm * (1.f / NPTS));
}

// ---------------------------------------------------------------------------
// final: out[32][40] = h2[32][256] @ wl3^T + bl3   (f32)
// ---------------------------------------------------------------------------
__global__ void final_fc_kernel(const float* __restrict__ h,
                                const float* __restrict__ w,
                                const float* __restrict__ bias,
                                float* __restrict__ out) {
  int i = blockIdx.x * 128 + threadIdx.x;
  if (i >= 32 * 40) return;
  int b = i / 40, o = i - b * 40;
  float acc = bias[o];
  const float* hb = h + (size_t)b * 256;
  const float* wo = w + (size_t)o * 256;
  for (int c = 0; c < 256; ++c) acc += hb[c] * wo[c];
  out[i] = acc;
}

// ---------------------------------------------------------------------------
// host side
// ---------------------------------------------------------------------------
static inline unsigned cdiv(long a, long b) { return (unsigned)((a + b - 1) / b); }

extern "C" void kernel_launch(void* const* d_in, const int* in_sizes, int n_in,
                              void* d_out, int out_size, void* d_ws, size_t ws_size,
                              hipStream_t stream) {
  (void)in_sizes; (void)n_in; (void)out_size; (void)ws_size;

  const float* x   = (const float*)d_in[0];
  const float* w1  = (const float*)d_in[1];
  const float* bn1 = (const float*)d_in[2];
  const float* w2  = (const float*)d_in[3];
  const float* bn2 = (const float*)d_in[4];
  const float* w3  = (const float*)d_in[5];
  const float* bn3 = (const float*)d_in[6];
  const float* w4  = (const float*)d_in[7];
  const float* bn4 = (const float*)d_in[8];
  const float* w5  = (const float*)d_in[9];
  const float* bn5 = (const float*)d_in[10];
  const float* wl1 = (const float*)d_in[11];
  const float* bn6 = (const float*)d_in[12];
  const float* wl2 = (const float*)d_in[13];
  const float* bl2 = (const float*)d_in[14];
  const float* bn7 = (const float*)d_in[15];
  const float* wl3 = (const float*)d_in[16];
  const float* bl3 = (const float*)d_in[17];
  float* out = (float*)d_out;

  const long BN = (long)BATCH * NPTS;
  char* ws = (char*)d_ws;
  size_t off = 0;
  auto alloc = [&](size_t bytes) {
    size_t o = off;
    off = (off + bytes + 255) & ~size_t(255);
    return o;
  };

  size_t o_pts   = alloc(BN * 3 * sizeof(float));
  size_t o_idx   = alloc(BN * KNB * sizeof(int));
  size_t o_canon = alloc(BN * KNB * 3 * sizeof(float));
  size_t o_perm  = alloc(BN * KNB * sizeof(int));
  // big region: reused as (a) patch features P [BN][2624] f16,
  // (b) gram/dist [B][N][N] f32 (134MB), (c) x5 [BN][1024] f32 (134MB)
  size_t o_P     = alloc(BN * 2624 * sizeof(_Float16));
  size_t o_x1    = alloc(BN * 64  * sizeof(float));
  size_t o_x2    = alloc(BN * 64  * sizeof(float));
  size_t o_x3    = alloc(BN * 128 * sizeof(float));
  size_t o_x4    = alloc(BN * 256 * sizeof(float));
  size_t o_x1h   = alloc(BN * 64  * sizeof(_Float16));
  size_t o_x2h   = alloc(BN * 64  * sizeof(_Float16));
  size_t o_x3h   = alloc(BN * 128 * sizeof(_Float16));
  size_t o_diag  = alloc(BN * sizeof(float));
  size_t o_xc    = alloc(BN * 512 * sizeof(_Float16));
  size_t o_xph   = alloc((size_t)BATCH * 2048 * sizeof(_Float16));
  size_t o_h1    = alloc((size_t)BATCH * 512 * sizeof(float));
  size_t o_h1h   = alloc((size_t)BATCH * 512 * sizeof(_Float16));
  size_t o_h2    = alloc((size_t)BATCH * 256 * sizeof(float));
  size_t o_w1h   = alloc((size_t)64   * 64   * sizeof(_Float16));
  size_t o_w2h   = alloc((size_t)64   * 1344 * sizeof(_Float16));
  size_t o_w3h   = alloc((size_t)128  * 1344 * sizeof(_Float16));
  size_t o_w4h   = alloc((size_t)256  * 2624 * sizeof(_Float16));
  size_t o_w5h   = alloc((size_t)1024 * 512  * sizeof(_Float16));
  size_t o_wl1h  = alloc((size_t)512  * 2048 * sizeof(_Float16));
  size_t o_wl2h  = alloc((size_t)256  * 512  * sizeof(_Float16));
  size_t o_sc[7], o_sh[7];
  const int bnC[7] = {64, 64, 128, 256, 1024, 512, 256};
  for (int i = 0; i < 7; ++i) {
    o_sc[i] = alloc(bnC[i] * sizeof(float));
    o_sh[i] = alloc(bnC[i] * sizeof(float));
  }

  float*    pts   = (float*)(ws + o_pts);
  int*      idx   = (int*)(ws + o_idx);
  float*    canon = (float*)(ws + o_canon);
  int*      perm  = (int*)(ws + o_perm);
  _Float16* P     = (_Float16*)(ws + o_P);
  float*    dist  = (float*)(ws + o_P);     // alias (P dead during knn)
  float*    x5    = (float*)(ws + o_P);     // alias (P dead after conv4)
  float*    x1o   = (float*)(ws + o_x1);
  float*    x2o   = (float*)(ws + o_x2);
  float*    x3o   = (float*)(ws + o_x3);
  float*    x4o   = (float*)(ws + o_x4);
  _Float16* x1h   = (_Float16*)(ws + o_x1h);
  _Float16* x2h   = (_Float16*)(ws + o_x2h);
  _Float16* x3h   = (_Float16*)(ws + o_x3h);
  float*    diag  = (float*)(ws + o_diag);
  _Float16* xc    = (_Float16*)(ws + o_xc);
  _Float16* xph   = (_Float16*)(ws + o_xph);
  float*    h1    = (float*)(ws + o_h1);
  _Float16* h1h   = (_Float16*)(ws + o_h1h);
  float*    h2    = (float*)(ws + o_h2);
  _Float16* w1h   = (_Float16*)(ws + o_w1h);
  _Float16* w2h   = (_Float16*)(ws + o_w2h);
  _Float16* w3h   = (_Float16*)(ws + o_w3h);
  _Float16* w4h   = (_Float16*)(ws + o_w4h);
  _Float16* w5h   = (_Float16*)(ws + o_w5h);
  _Float16* wl1h  = (_Float16*)(ws + o_wl1h);
  _Float16* wl2h  = (_Float16*)(ws + o_wl2h);

  // ---- parameter prep -----------------------------------------------------
  const float* bns[7]  = {bn1, bn2, bn3, bn4, bn5, bn6, bn7};
  const float* bias[7] = {nullptr, nullptr, nullptr, nullptr, nullptr, nullptr, bl2};
  for (int i = 0; i < 7; ++i) {
    bn_prep_kernel<<<cdiv(bnC[i], 256), 256, 0, stream>>>(
        bns[i], bias[i], (float*)(ws + o_sc[i]), (float*)(ws + o_sh[i]), bnC[i]);
  }
  struct { const float* w; _Float16* wh; int Cc, Cp; long total; } wc[7] = {
      {w1,  w1h,  60,   64,   (long)64   * 64},
      {w2,  w2h,  1340, 1344, (long)64   * 1344},
      {w3,  w3h,  1340, 1344, (long)128  * 1344},
      {w4,  w4h,  2620, 2624, (long)256  * 2624},
      {w5,  w5h,  512,  512,  (long)1024 * 512},
      {wl1, wl1h, 2048, 2048, (long)512  * 2048},
      {wl2, wl2h, 512,  512,  (long)256  * 512}};
  for (int i = 0; i < 7; ++i)
    wconv_kernel<<<cdiv(wc[i].total, 256), 256, 0, stream>>>(
        wc[i].w, wc[i].wh, wc[i].Cc, wc[i].Cp, wc[i].total);

  // ---- global canonicalization -------------------------------------------
  global_pca_kernel<<<BATCH, 512, 0, stream>>>(x, pts);

  auto gemm_conv = [&](const _Float16* Wm, const _Float16* Xm, float* Y,
                       _Float16* Yh, int bni, int O, int Cp) {
    dim3 grid((O / 64) * (NPTS / 128), BATCH);
    gemm_wmma_bn_blk_kernel<<<grid, 256, 0, stream>>>(
        Wm, Xm, Y, Yh, (float*)(ws + o_sc[bni]), (float*)(ws + o_sh[bni]),
        O, Cp, NPTS);
  };
  auto gemm_fc = [&](const _Float16* Wm, const _Float16* Xm, float* Y, int bni,
                     int O, int Cp, int Np) {
    int ntiles = (O / 16) * (Np / 16);
    dim3 grid(cdiv(ntiles, 8), 1);
    gemm_wmma_bn_kernel<<<grid, 256, 0, stream>>>(
        Wm, Xm, Y, (float*)(ws + o_sc[bni]), (float*)(ws + o_sh[bni]), O, Cp, Np);
  };
  auto knn_wmma = [&](const _Float16* Fh, int Cc) {
    dim3 g1((NPTS / 64) * (NPTS / 128), BATCH);
    gram_wmma_kernel<<<g1, 256, 0, stream>>>(Fh, dist, Cc);
    extract_diag_kernel<<<cdiv(BN, 256), 256, 0, stream>>>(dist, diag);
    topk_gram_kernel<<<dim3(NPTS / 64, BATCH), 64, 0, stream>>>(dist, diag, idx);
  };

  unsigned patch_blocks = (unsigned)(BN / 8);

  // ---- stage 1 (C=3: scalar knn) -----------------------------------------
  knn_topk_kernel<<<dim3(NPTS / 64, BATCH), 64, 0, stream>>>(pts, idx, 3);
  patch_pca_kernel<<<patch_blocks, 256, 0, stream>>>(pts, idx, canon, perm);
  build_p1_kernel<<<cdiv(BN * 64, 256), 256, 0, stream>>>(canon, P);
  gemm_conv(w1h, P, x1o, x1h, 0, 64, 64);

  // ---- stage 2 (knn via WMMA gram) ---------------------------------------
  knn_wmma(x1h, 64);
  patch_pca_kernel<<<patch_blocks, 256, 0, stream>>>(pts, idx, canon, perm);
  build_pf_kernel<<<(unsigned)BN, 256, 0, stream>>>(canon, idx, perm, x1o, P, 64, 1344);
  gemm_conv(w2h, P, x2o, x2h, 1, 64, 1344);

  // ---- stage 3 ------------------------------------------------------------
  knn_wmma(x2h, 64);
  patch_pca_kernel<<<patch_blocks, 256, 0, stream>>>(pts, idx, canon, perm);
  build_pf_kernel<<<(unsigned)BN, 256, 0, stream>>>(canon, idx, perm, x2o, P, 64, 1344);
  gemm_conv(w3h, P, x3o, x3h, 2, 128, 1344);

  // ---- stage 4 ------------------------------------------------------------
  knn_wmma(x3h, 128);
  patch_pca_kernel<<<patch_blocks, 256, 0, stream>>>(pts, idx, canon, perm);
  build_pf_kernel<<<(unsigned)BN, 256, 0, stream>>>(canon, idx, perm, x3o, P, 128, 2624);
  gemm_conv(w4h, P, x4o, nullptr, 3, 256, 2624);

  // ---- conv5 + pooling ----------------------------------------------------
  build_xc_kernel<<<cdiv(BN * 512, 256), 256, 0, stream>>>(x1o, x2o, x3o, x4o, xc);
  gemm_conv(w5h, xc, x5, nullptr, 4, 1024, 512);   // x5 aliases P (dead)
  pool_kernel<<<dim3(1024 / 256, BATCH), 256, 0, stream>>>(x5, xph);

  // ---- FC head ------------------------------------------------------------
  gemm_fc(wl1h, xph, h1, 5, 512, 2048, 32);
  cvt_f32_to_f16<<<cdiv((long)BATCH * 512, 256), 256, 0, stream>>>(h1, h1h, (long)BATCH * 512);
  gemm_fc(wl2h, h1h, h2, 6, 256, 512, 32);
  final_fc_kernel<<<cdiv(32 * 40, 128), 128, 0, stream>>>(h2, wl3, bl3, out);
}
